// ConfGNN_20117626814605
// MI455X (gfx1250) — matrix operand order repaired and checked by reading, not compiled, not run on hardware
//
#include <hip/hip_runtime.h>

typedef __attribute__((ext_vector_type(16))) _Float16 v16h;
typedef __attribute__((ext_vector_type(8)))  float    v8f;

#define N_FEAT 500
#define N_CLS  40
#define HID    64
#define KPAD   512   // 500 padded to 16 chunks of K=32
#define NTILE  48    // 40 cols padded to 3 x 16
#define ALPHA  0.1f

// ---------------------------------------------------------------- utilities
__global__ void fill_zero(float* __restrict__ p, int n) {
  int i = blockIdx.x * blockDim.x + threadIdx.x;
  int stride = gridDim.x * blockDim.x;
  for (; i < n; i += stride) p[i] = 0.f;
}

// W_base [500,40] f32 row-major  ->  Bp [48 cols][512 k] f16, k-contiguous, zero padded
__global__ void pack_wbase(const float* __restrict__ Wb, _Float16* __restrict__ Bp) {
  int idx = blockIdx.x * blockDim.x + threadIdx.x;
  if (idx >= NTILE * KPAD) return;
  int c = idx >> 9, k = idx & (KPAD - 1);
  float v = (k < N_FEAT && c < N_CLS) ? Wb[k * N_CLS + c] : 0.f;
  Bp[c * KPAD + k] = (_Float16)v;
}

// ------------------------------------------- fused scorer + softmax + MLP
__global__ __launch_bounds__(128) void fused_scorer(
    const float* __restrict__ x, const _Float16* __restrict__ Bp,
    const float* __restrict__ b_base,
    const float* __restrict__ W1, const float* __restrict__ b1,
    const float* __restrict__ W2, const float* __restrict__ b2,
    float* __restrict__ scores_out, float* __restrict__ h0_out, int n_nodes)
{
  __shared__ float sW1[N_CLS * HID];
  __shared__ float sW2[HID * N_CLS];
  __shared__ float sB1[HID];
  __shared__ float sB2[N_CLS];
  __shared__ float sB0[NTILE];
  __shared__ float sProb[4][16][NTILE];
  __shared__ float sH1[4][16][HID];

  int tid = threadIdx.x;
  for (int i = tid; i < N_CLS * HID; i += 128) sW1[i] = W1[i];
  for (int i = tid; i < HID * N_CLS; i += 128) sW2[i] = W2[i];
  if (tid < HID)   sB1[tid] = b1[tid];
  if (tid < N_CLS) sB2[tid] = b2[tid];
  if (tid < NTILE) sB0[tid] = (tid < N_CLS) ? b_base[tid] : 0.f;
  __syncthreads();

  int wid  = tid >> 5;
  int lane = tid & 31;
  int hi   = (lane >> 4) & 1;         // half-wave select
  int colB = lane & 15;
  int m0   = (blockIdx.x * 4 + wid) * 16;
  int row  = m0 + (lane & 15);
  int rowc = row < n_nodes ? row : (n_nodes - 1);
  const float* xr = x + (size_t)rowc * N_FEAT;

  // B fragments: lanes 0-15 hold K=kb..kb+15, lanes 16-31 hold K=kb+16..kb+31
  const uint32_t* bp0 = (const uint32_t*)(Bp + (size_t)(0 * 16 + colB) * KPAD);
  const uint32_t* bp1 = (const uint32_t*)(Bp + (size_t)(1 * 16 + colB) * KPAD);
  const uint32_t* bp2 = (const uint32_t*)(Bp + (size_t)(2 * 16 + colB) * KPAD);

  v8f acc0 = {}, acc1 = {}, acc2 = {};
  int kbA = hi * 16;                  // A: lanes16-31 start at K+16
  int kbB = hi * 16;                  // B: lanes16-31 start at K+16

  // --- 15 full K-chunks: no bounds checks, clean b64/b128 loads + cvt_pk ---
  for (int kc = 0; kc < 15; ++kc) {
    int kk = kc * 32;
    v16h a;
#pragma unroll
    for (int j = 0; j < 8; ++j) {
      // A contiguous K layout: VGPR j -> K=kb+2j..2j+1 (matches B)
      int k0 = kk + kbA + 2 * j;
      float2 p = *(const float2*)(xr + k0);   // k0 even -> 8B aligned
      a[2 * j]     = (_Float16)p.x;
      a[2 * j + 1] = (_Float16)p.y;
    }
    union { v16h h; uint32_t u[8]; } fb0, fb1, fb2;
    int kof = (kk + kbB) >> 1;
#pragma unroll
    for (int j = 0; j < 8; ++j) {
      fb0.u[j] = bp0[kof + j];
      fb1.u[j] = bp1[kof + j];
      fb2.u[j] = bp2[kof + j];
    }
    acc0 = __builtin_amdgcn_wmma_f32_16x16x32_f16(false, a, false, fb0.h, (short)0, acc0, false, false);
    acc1 = __builtin_amdgcn_wmma_f32_16x16x32_f16(false, a, false, fb1.h, (short)0, acc1, false, false);
    acc2 = __builtin_amdgcn_wmma_f32_16x16x32_f16(false, a, false, fb2.h, (short)0, acc2, false, false);
  }

  // --- guarded epilogue chunk: k = 480..511, valid only below 500 ---
  {
    const int kk = 480;
    v16h a;
#pragma unroll
    for (int j = 0; j < 8; ++j) {
      int k0 = kk + kbA + 2 * j;
      float f0 = (k0     < N_FEAT) ? xr[k0]     : 0.f;
      float f1 = (k0 + 1 < N_FEAT) ? xr[k0 + 1] : 0.f;
      a[2 * j]     = (_Float16)f0;
      a[2 * j + 1] = (_Float16)f1;
    }
    union { v16h h; uint32_t u[8]; } fb0, fb1, fb2;
    int kof = (kk + kbB) >> 1;
#pragma unroll
    for (int j = 0; j < 8; ++j) {
      fb0.u[j] = bp0[kof + j];
      fb1.u[j] = bp1[kof + j];
      fb2.u[j] = bp2[kof + j];
    }
    acc0 = __builtin_amdgcn_wmma_f32_16x16x32_f16(false, a, false, fb0.h, (short)0, acc0, false, false);
    acc1 = __builtin_amdgcn_wmma_f32_16x16x32_f16(false, a, false, fb1.h, (short)0, acc1, false, false);
    acc2 = __builtin_amdgcn_wmma_f32_16x16x32_f16(false, a, false, fb2.h, (short)0, acc2, false, false);
  }

  // C layout: VGPR j -> row (j + hi*8), col (lane&15) + 16*t
#pragma unroll
  for (int j = 0; j < 8; ++j) {
    int r    = j + hi * 8;
    int grow = m0 + r;
    float v0 = acc0[j] + sB0[colB];
    float v1 = acc1[j] + sB0[colB + 16];
    float v2 = acc2[j] + sB0[colB + 32];
    sProb[wid][r][colB]      = v0;
    sProb[wid][r][colB + 16] = v1;
    sProb[wid][r][colB + 32] = v2;
    if (grow < n_nodes) {
      size_t base = (size_t)grow * N_CLS;
      scores_out[base + colB]      = v0;
      scores_out[base + colB + 16] = v1;
      if (colB + 32 < N_CLS) scores_out[base + colB + 32] = v2;
    }
  }
  __syncthreads();

  // Row softmax: one lane per row
  if (lane < 16) {
    float* p = sProb[wid][lane];
    float mx = p[0];
    for (int c = 1; c < N_CLS; ++c) mx = fmaxf(mx, p[c]);
    float s = 0.f;
    for (int c = 0; c < N_CLS; ++c) { float e = __expf(p[c] - mx); p[c] = e; s += e; }
    float inv = 1.f / s;
    for (int c = 0; c < N_CLS; ++c) p[c] *= inv;
  }
  __syncthreads();

  // h1 = relu(probs @ W1 + b1)
  for (int idx = lane; idx < 16 * HID; idx += 32) {
    int r = idx >> 6, c = idx & (HID - 1);
    float acc = sB1[c];
    for (int kk2 = 0; kk2 < N_CLS; ++kk2)
      acc = fmaf(sProb[wid][r][kk2], sW1[kk2 * HID + c], acc);
    sH1[wid][r][c] = fmaxf(acc, 0.f);
  }
  __syncthreads();

  // h0 = h1 @ W2 + b2
  for (int idx = lane; idx < 16 * N_CLS; idx += 32) {
    int r = idx / N_CLS, c = idx - r * N_CLS;
    float acc = sB2[c];
    for (int kk2 = 0; kk2 < HID; ++kk2)
      acc = fmaf(sH1[wid][r][kk2], sW2[kk2 * N_CLS + c], acc);
    int grow = m0 + r;
    if (grow < n_nodes) h0_out[(size_t)grow * N_CLS + c] = acc;
  }
}

// ---------------------------------------------------------------- APPNP
__global__ void deg_count(const int* __restrict__ src, const int* __restrict__ dst,
                          float* __restrict__ degO, float* __restrict__ degI, int E) {
  int e = blockIdx.x * blockDim.x + threadIdx.x;
  if (e >= E) return;
  atomicAdd(&degO[src[e]], 1.f);
  atomicAdd(&degI[dst[e]], 1.f);
}

__global__ void edge_norm(const int* __restrict__ src, const int* __restrict__ dst,
                          const float* __restrict__ degO, const float* __restrict__ degI,
                          float* __restrict__ nrm, int E) {
  int e = blockIdx.x * blockDim.x + threadIdx.x;
  if (e >= E) return;
  float dO = degO[src[e]];
  float dI = degI[dst[e]];
  float a = dO > 0.f ? rsqrtf(dO) : 0.f;
  float b = dI > 0.f ? rsqrtf(dI) : 0.f;
  nrm[e] = a * b;
}

// agg[dst] += h[src] * norm   (one thread per edge; h and agg are L2-resident)
__global__ void scatter_add(const int* __restrict__ src, const int* __restrict__ dst,
                            const float* __restrict__ nrm, const float* __restrict__ h,
                            float* __restrict__ agg, int E) {
  int e = blockIdx.x * blockDim.x + threadIdx.x;
  if (e >= E) return;
  int s = src[e], d = dst[e];
  float w = nrm[e];
  const float4* hs = (const float4*)(h + (size_t)s * N_CLS);
  float* ad = agg + (size_t)d * N_CLS;
#pragma unroll
  for (int i = 0; i < N_CLS / 4; ++i) {
    float4 v = hs[i];
    atomicAdd(&ad[4 * i + 0], v.x * w);
    atomicAdd(&ad[4 * i + 1], v.y * w);
    atomicAdd(&ad[4 * i + 2], v.z * w);
    atomicAdd(&ad[4 * i + 3], v.w * w);
  }
}

// out = (1-a)*agg + a*h0 ; re-zero agg for the next hop
__global__ void combine(float* __restrict__ agg, const float* __restrict__ h0,
                        float* __restrict__ out, int n) {
  int i = blockIdx.x * blockDim.x + threadIdx.x;
  if (i >= n) return;
  out[i] = (1.f - ALPHA) * agg[i] + ALPHA * h0[i];
  agg[i] = 0.f;
}

// ---------------------------------------------------------------- launcher
// ws layout (floats): h0[NC] | hA[NC] | agg[NC] | degO[n] | degI[n] | norm[E] | Bp(f16 48*512)
// ~55 MB required. d_out's adjust region doubles as the odd-hop ping-pong buffer.
extern "C" void kernel_launch(void* const* d_in, const int* in_sizes, int n_in,
                              void* d_out, int out_size, void* d_ws, size_t ws_size,
                              hipStream_t stream) {
  const float* x  = (const float*)d_in[0];
  const int*   ei = (const int*)  d_in[1];
  const float* Wb = (const float*)d_in[2];
  const float* bb = (const float*)d_in[3];
  const float* W1 = (const float*)d_in[4];
  const float* b1 = (const float*)d_in[5];
  const float* W2 = (const float*)d_in[6];
  const float* b2 = (const float*)d_in[7];

  int n = in_sizes[0] / N_FEAT;     // 100000
  int E = in_sizes[1] / 2;          // 1600000
  size_t NC = (size_t)n * N_CLS;

  float* ws   = (float*)d_ws;
  float* h0   = ws;
  float* hA   = h0 + NC;
  float* agg  = hA + NC;
  float* degO = agg + NC;
  float* degI = degO + n;
  float* nrm  = degI + n;
  _Float16* Bp = (_Float16*)(nrm + E);

  float* adj_out    = (float*)d_out;         // also serves as odd-hop buffer
  float* scores_out = adj_out + NC;

  const int* srcI = ei;
  const int* dstI = ei + E;

  fill_zero<<<1024, 256, 0, stream>>>(agg, (int)NC);
  fill_zero<<<64, 256, 0, stream>>>(degO, 2 * n);
  pack_wbase<<<(NTILE * KPAD + 255) / 256, 256, 0, stream>>>(Wb, Bp);

  int tiles  = (n + 15) / 16;
  int blocks = (tiles + 3) / 4;
  fused_scorer<<<blocks, 128, 0, stream>>>(x, Bp, bb, W1, b1, W2, b2,
                                           scores_out, h0, n);

  int eb = (E + 255) / 256;
  deg_count<<<eb, 256, 0, stream>>>(srcI, dstI, degO, degI, E);
  edge_norm<<<eb, 256, 0, stream>>>(srcI, dstI, degO, degI, nrm, E);

  const float* cur = h0;
  int nb = ((int)NC + 255) / 256;
  for (int hop = 0; hop < 10; ++hop) {
    scatter_add<<<eb, 256, 0, stream>>>(srcI, dstI, nrm, cur, agg, E);
    float* nxt = (hop & 1) ? adj_out : hA;   // hop 9 (odd) lands in d_out
    combine<<<nb, 256, 0, stream>>>(agg, h0, nxt, (int)NC);
    cur = nxt;
  }
}
